// DeformableConv2d_30219389894746
// MI455X (gfx1250) — matrix operand order repaired and checked by reading, not compiled, not run on hardware
//
#include <hip/hip_runtime.h>
#include <hip/hip_bf16.h>

typedef __attribute__((ext_vector_type(16))) _Float16 v16h;
typedef __attribute__((ext_vector_type(8)))  float    v8f;
typedef __attribute__((ext_vector_type(4)))  float    v4f;
typedef __attribute__((ext_vector_type(2)))  float    v2f;

#define BDIM 8
#define HDIM 128
#define WDIM 128
#define CDIM 64
#define FDIM 64
#define NTAP 9
#define PIX_PER_BLOCK 128       // 8 waves * 16 pixels
#define THREADS 256
#define TILES_PER_BLOCK 4       // persistent: each block handles 4 pixel tiles
#define NBLOCKS 256             // 256 * 4 * 128 = 131072 pixels

// W staged in LDS as f16, pre-swizzled into WMMA B-fragment order:
// fragment (tap, kchunk, ntile): 32 lanes x 16 halfs contiguous.
// lane -> N = lane&15 (+ntile*16); holds K = 16*(lane>>4) + j, j=0..15
// (global k = tap*64 + kchunk*32 + K  -> W[tap][c][f])
#define NFRAG (NTAP * 2 * 4)           // 72 fragments
#define LDSW_HALFS (NFRAG * 512)       // 36864 halfs = 73728 B

__global__ __launch_bounds__(THREADS)
void deform_conv_wmma_kernel(const float* __restrict__ xin,
                             const float* __restrict__ offs,
                             const float* __restrict__ Wt,
                             const float* __restrict__ bias,
                             float* __restrict__ out) {
    __shared__ _Float16 ldsW[LDSW_HALFS];

    const int tid = threadIdx.x;

    // ---- stage W (f32 -> f16) into fragment-linear LDS layout ----
    // Source-linear: each thread does 36 independent b128 loads of W,
    // then scatters 4 halfs into the swizzled fragment layout.
#pragma unroll 6
    for (int i = 0; i < (NTAP * CDIM * FDIM) / (THREADS * 4); ++i) {  // 36
        const int e  = ((i << 8) + tid) << 2;       // linear f32 index into W
        const v4f w4 = *(const v4f*)(Wt + e);
        const int tap = e >> 12;                    // / (64*64)
        const int c   = (e >> 6) & 63;
        const int fb  = e & 63;                     // f base (multiple of 4)
        const int kchunk = c >> 5;
        const int khb    = (c >> 4) & 1;
        const int j      = c & 15;
        const int frag   = (tap << 3) + (kchunk << 2) + (fb >> 4);
        const int dbase  = (frag << 9) + (((khb << 4) + (fb & 15)) << 4) + j;
        ldsW[dbase +  0] = (_Float16)w4.x;
        ldsW[dbase + 16] = (_Float16)w4.y;
        ldsW[dbase + 32] = (_Float16)w4.z;
        ldsW[dbase + 48] = (_Float16)w4.w;
    }
    __syncthreads();

    // ---- per-wave tile setup (wave32) ----
    const int lane  = tid & 31;
    const int wave  = tid >> 5;
    const int Mrow  = lane & 15;     // pixel row of A / output tile
    const int khalf = lane >> 4;     // which K-half this lane supplies for A
    const int fcol  = lane & 15;
    const int c0    = khalf << 4;    // channel base offset from lane's K-half

    float bv[4];
#pragma unroll
    for (int nt = 0; nt < 4; ++nt) bv[nt] = bias[(nt << 4) + fcol];

    // ---- persistent loop over pixel tiles (W stays resident in LDS) ----
    for (int it = 0; it < TILES_PER_BLOCK; ++it) {
        const int pixWaveBase =
            (blockIdx.x * TILES_PER_BLOCK + it) * PIX_PER_BLOCK + (wave << 4);
        const int pix  = pixWaveBase + Mrow;
        const int xcol = pix & (WDIM - 1);
        const int yrow = (pix >> 7) & (HDIM - 1);
        const int bidx = pix >> 14;
        const int rowb = bidx * HDIM;

        v8f acc[4];
#pragma unroll
        for (int nt = 0; nt < 4; ++nt)
            acc[nt] = (v8f){0.f,0.f,0.f,0.f,0.f,0.f,0.f,0.f};

#pragma unroll
        for (int tap = 0; tap < NTAP; ++tap) {
            const int ti = tap / 3 - 1;
            const int tj = tap % 3 - 1;

            // sampling coordinates (clipped), exactly as reference
            const v2f o2 = *(const v2f*)(offs + pix * (2 * NTAP) + (tap << 1));
            float cy = (float)(yrow + ti) + o2.x;
            float cx = (float)(xcol + tj) + o2.y;
            cy = fminf(fmaxf(cy, 0.f), (float)(HDIM - 1));
            cx = fminf(fmaxf(cx, 0.f), (float)(WDIM - 1));
            const int y0 = (int)floorf(cy);
            const int y1 = (int)ceilf(cy);
            const int x0 = (int)floorf(cx);
            const int x1 = (int)ceilf(cx);
            const float fy = cy - (float)y0;
            const float fx = cx - (float)x0;
            const float w00 = (1.f - fy) * (1.f - fx);
            const float w10 = fy * (1.f - fx);
            const float w01 = (1.f - fy) * fx;
            const float w11 = fy * fx;

            const float* p00 = xin + (((rowb + y0) * WDIM + x0) * CDIM);
            const float* p10 = xin + (((rowb + y1) * WDIM + x0) * CDIM);
            const float* p01 = xin + (((rowb + y0) * WDIM + x1) * CDIM);
            const float* p11 = xin + (((rowb + y1) * WDIM + x1) * CDIM);

#pragma unroll
            for (int kchunk = 0; kchunk < 2; ++kchunk) {
                // ---- build A fragment (16x32 f16), contiguous k = 16*khalf + s as in B ----
                v16h a;
#pragma unroll
                for (int r = 0; r < 2; ++r) {
                    const int cb = ((kchunk << 2) + r) * 8 + c0; // 32B aligned
                    const v4f* q00 = (const v4f*)(p00 + cb);
                    const v4f* q10 = (const v4f*)(p10 + cb);
                    const v4f* q01 = (const v4f*)(p01 + cb);
                    const v4f* q11 = (const v4f*)(p11 + cb);
#pragma unroll
                    for (int h = 0; h < 2; ++h) {
                        const v4f a00 = q00[h];
                        const v4f a10 = q10[h];
                        const v4f a01 = q01[h];
                        const v4f a11 = q11[h];
#pragma unroll
                        for (int e = 0; e < 4; ++e) {
                            const float s = w00 * a00[e] + w10 * a10[e]
                                          + w01 * a01[e] + w11 * a11[e];
                            a[(r << 3) + (h << 2) + e] = (_Float16)s;
                        }
                    }
                }

                // ---- 4 N-tiles of WMMA, B fragments streamed from LDS ----
                const int fragBase = (tap << 3) + (kchunk << 2);
#pragma unroll
                for (int nt = 0; nt < 4; ++nt) {
                    const v16h bfrag =
                        *(const v16h*)(ldsW + ((fragBase + nt) << 9) + (lane << 4));
                    acc[nt] = __builtin_amdgcn_wmma_f32_16x16x32_f16(
                        false, a, false, bfrag, (short)0, acc[nt], false, false);
                }
            }
        }

        // ---- store: VGPR r -> M = r + 8*khalf ; N = lane&15 ----
#pragma unroll
        for (int nt = 0; nt < 4; ++nt) {
#pragma unroll
            for (int r = 0; r < 8; ++r) {
                const int prow = pixWaveBase + (khalf << 3) + r;
                out[prow * FDIM + (nt << 4) + fcol] = acc[nt][r] + bv[nt];
            }
        }
    }
}

extern "C" void kernel_launch(void* const* d_in, const int* in_sizes, int n_in,
                              void* d_out, int out_size, void* d_ws, size_t ws_size,
                              hipStream_t stream) {
    (void)in_sizes; (void)n_in; (void)d_ws; (void)ws_size; (void)out_size;
    const float* x    = (const float*)d_in[0];
    const float* offs = (const float*)d_in[1];
    const float* Wt   = (const float*)d_in[2];
    const float* bias = (const float*)d_in[3];
    float* out        = (float*)d_out;

    deform_conv_wmma_kernel<<<NBLOCKS, THREADS, 0, stream>>>(x, offs, Wt, bias, out);
}
